// FocusedLinearAttention_18674517803387
// MI455X (gfx1250) — matrix operand order repaired and checked by reading, not compiled, not run on hardware
//
#include <hip/hip_runtime.h>

typedef unsigned short u16;
typedef unsigned int   u32;
typedef __bf16 bf16;
typedef bf16  v16bf __attribute__((ext_vector_type(16)));
typedef float v8f   __attribute__((ext_vector_type(8)));

#define EPS_ 1e-6f
// B=256, N=196, C=768, H=12, HD=64, S=14; M_TOTAL = 50176

__device__ __forceinline__ u16 f2b(float f) {            // fp32 -> bf16 (RNE)
  u32 u = __builtin_bit_cast(u32, f);
  u32 r = u + 0x7FFFu + ((u >> 16) & 1u);
  return (u16)(r >> 16);
}
__device__ __forceinline__ float b2f(u16 s) {
  u32 u = ((u32)s) << 16;
  return __builtin_bit_cast(float, u);
}
__device__ __forceinline__ u32 pack2(float lo, float hi) {
  return (u32)f2b(lo) | ((u32)f2b(hi) << 16);
}

union Frag { v16bf v; u16 s[16]; u32 u[8]; };

__device__ __forceinline__ v8f wmma_bf16(const Frag& a, const Frag& b, v8f c) {
  return __builtin_amdgcn_wmma_f32_16x16x32_bf16(false, a.v, false, b.v,
                                                 (short)0, c, false, false);
}

// ---------------------------------------------------------------------------
// Kernel A: Y(50176 x 2304) = x(50176x768) @ [Wq|Wk|Wv]^T  + fused epilogue
// block: 128(M) x 64(N), 8 waves -> each wave 32x32 (2x2 wmma tiles), K=768
// Double-buffered LDS, float4 global loads, packed bf16 LDS stores.
// ---------------------------------------------------------------------------
__global__ __launch_bounds__(256) void k_gemm_qkv(
    const float* __restrict__ x,
    const float* __restrict__ Wq, const float* __restrict__ bq,
    const float* __restrict__ Wk, const float* __restrict__ bk,
    const float* __restrict__ Wv, const float* __restrict__ bv,
    const float* __restrict__ scale_p, const float* __restrict__ pos,
    u16* __restrict__ qb, u16* __restrict__ kb, u16* __restrict__ vh)
{
  __shared__ __align__(16) u16 lA[2][128 * 32];   // [m][k] bf16
  __shared__ __align__(16) u16 lB[2][64 * 32];    // [n][k] bf16 (W row-major)

  const int t = threadIdx.x;
  const int w = t >> 5, lane = t & 31, half = lane >> 4, ln = lane & 15;
  const int wm = w >> 1, wn = w & 1;
  const int mblk = blockIdx.x * 128;
  const int nblk = blockIdx.y;                  // 0..35
  const int region = nblk / 12;                 // 0=q, 1=k, 2=v
  const int nloc = (nblk % 12) * 64;
  const float* W = (region == 0) ? Wq : ((region == 1) ? Wk : Wv);

  auto load_tiles = [&](int kk, int buf) {
    const int k0 = kk * 32;
    for (int idx = t; idx < 1024; idx += 256) {          // 128x32 / 4
      int r = idx >> 3, c = (idx & 7) << 2;
      float4 f = *(const float4*)&x[(size_t)(mblk + r) * 768 + k0 + c];
      *(uint2*)&lA[buf][r * 32 + c] = make_uint2(pack2(f.x, f.y), pack2(f.z, f.w));
    }
    for (int idx = t; idx < 512; idx += 256) {           // 64x32 / 4
      int n = idx >> 3, c = (idx & 7) << 2;
      float4 f = *(const float4*)&W[(size_t)(nloc + n) * 768 + k0 + c];
      *(uint2*)&lB[buf][n * 32 + c] = make_uint2(pack2(f.x, f.y), pack2(f.z, f.w));
    }
  };

  v8f zero = {};
  v8f acc[2][2];
  for (int i = 0; i < 2; ++i) for (int j = 0; j < 2; ++j) acc[i][j] = zero;

  load_tiles(0, 0);
  for (int kk = 0; kk < 24; ++kk) {
    __syncthreads();
    if (kk + 1 < 24) load_tiles(kk + 1, (kk + 1) & 1);   // overlap with WMMA
    const u16* A_ = lA[kk & 1];
    const u16* B_ = lB[kk & 1];

    Frag a[2], b[2];
    for (int mt = 0; mt < 2; ++mt) {           // A frag: lanes->M, contig K runs
      int m = wm * 32 + mt * 16 + ln;
      int kbs = half * 16;
      for (int i = 0; i < 8; ++i)
        a[mt].u[i] = *(const u32*)&A_[m * 32 + kbs + 2 * i];
    }
    for (int nt = 0; nt < 2; ++nt) {           // B frag: lanes->N, contig K
      int n = wn * 32 + nt * 16 + ln;
      int kbs = half * 16;
      for (int i = 0; i < 8; ++i)
        b[nt].u[i] = *(const u32*)&B_[n * 32 + kbs + 2 * i];
    }
    for (int mt = 0; mt < 2; ++mt)
      for (int nt = 0; nt < 2; ++nt)
        acc[mt][nt] = wmma_bf16(a[mt], b[nt], acc[mt][nt]);
  }

  // Epilogue
  for (int mt = 0; mt < 2; ++mt)
    for (int nt = 0; nt < 2; ++nt)
      for (int r = 0; r < 8; ++r) {
        int m_g = mblk + wm * 32 + mt * 16 + r + half * 8;
        int c = nloc + wn * 32 + nt * 16 + ln;      // 0..767 within region
        float val = acc[mt][nt][r];
        if (region == 0) {
          val += bq[c];
          val = fmaxf(val, 0.f) + EPS_;
          val /= log1pf(expf(scale_p[c]));          // softplus
          qb[(size_t)m_g * 768 + c] = f2b(val);
        } else if (region == 1) {
          int tok = m_g % 196;
          val += bk[c] + pos[tok * 768 + c];
          val = fmaxf(val, 0.f) + EPS_;
          val /= log1pf(expf(scale_p[c]));
          kb[(size_t)m_g * 768 + c] = f2b(val);
        } else {
          val += bv[c];
          int bi = m_g / 196, tok = m_g % 196;
          int h = c >> 6, d = c & 63;
          vh[(size_t)((bi * 12 + h) * 196 + tok) * 64 + d] = f2b(val);
        }
      }
}

// ---------------------------------------------------------------------------
// Kernel B: focusing power + renorm over full C=768 row; emits head-split bf16
// ---------------------------------------------------------------------------
__global__ __launch_bounds__(256) void k_focus(
    const u16* __restrict__ qb, const u16* __restrict__ kb,
    u16* __restrict__ qh, u16* __restrict__ kh)
{
  __shared__ float red[4 * 256];
  const int r = blockIdx.x, t = threadIdx.x;
  float q3[3], k3[3];
  float s2q = 0.f, s6q = 0.f, s2k = 0.f, s6k = 0.f;
  for (int i = 0; i < 3; ++i) {
    int c = t + i * 256;
    float qv = b2f(qb[(size_t)r * 768 + c]);
    float kv = b2f(kb[(size_t)r * 768 + c]);
    s2q += qv * qv; s2k += kv * kv;
    q3[i] = qv * qv * qv; k3[i] = kv * kv * kv;
    s6q += q3[i] * q3[i]; s6k += k3[i] * k3[i];
  }
  red[t] = s2q; red[256 + t] = s6q; red[512 + t] = s2k; red[768 + t] = s6k;
  __syncthreads();
  for (int st = 128; st > 0; st >>= 1) {
    if (t < st)
      for (int j = 0; j < 4; ++j) red[j * 256 + t] += red[j * 256 + t + st];
    __syncthreads();
  }
  float fq = sqrtf(red[0]   / (red[256] + 1e-30f));   // ||q|| / ||q^3||
  float fk = sqrtf(red[512] / (red[768] + 1e-30f));
  int bi = r / 196, tok = r % 196;
  for (int i = 0; i < 3; ++i) {
    int c = t + i * 256;
    int h = c >> 6, d = c & 63;
    size_t dst = (size_t)((bi * 12 + h) * 196 + tok) * 64 + d;
    qh[dst] = f2b(q3[i] * fq);
    kh[dst] = f2b(k3[i] * fk);
  }
}

// ---------------------------------------------------------------------------
// Kernel C: per-head linear attention + depthwise conv; 1 block per (b,h)
//   kv(64x64) = k^T(64x196) @ v(196x64)   (WMMA; only last K-chunk guarded)
//   out(196x64) = q(196x64) @ kv(64x64)   (WMMA; only last M-tile guarded)
// ---------------------------------------------------------------------------
__global__ __launch_bounds__(256) void k_attn_head(
    const u16* __restrict__ qh, const u16* __restrict__ kh,
    const u16* __restrict__ vh,
    const float* __restrict__ dwc_w, const float* __restrict__ dwc_b,
    u16* __restrict__ attn)
{
  __shared__ __align__(16) u16 lk[196 * 64];   // k tile [j][c] (reused as ffm)
  __shared__ __align__(16) u16 lv[196 * 64];   // v tile [j][d]
  __shared__ __align__(16) u16 kvT[64 * 64];   // kv transposed [d2][c] bf16
  __shared__ float ksum[64];
  __shared__ float zr[196];
  u16* ffm = lk;                               // conv output reuses lk

  const int t = threadIdx.x;
  const int w = t >> 5, lane = t & 31, half = lane >> 4, ln = lane & 15;
  const int head = blockIdx.x;
  const int bi = head / 12, h = head % 12;
  const u16* qhd = qh + (size_t)head * 196 * 64;
  const u16* khd = kh + (size_t)head * 196 * 64;
  const u16* vhd = vh + (size_t)head * 196 * 64;

  for (int i = t; i < 1568; i += 256) {        // 196*64*2B / 16B = 1568
    ((uint4*)lk)[i] = ((const uint4*)khd)[i];
    ((uint4*)lv)[i] = ((const uint4*)vhd)[i];
  }
  __syncthreads();

  if (t < 64) {                                // ksum[d] = sum_j k[j][d]
    float s = 0.f;
    for (int j = 0; j < 196; ++j) s += b2f(lk[j * 64 + t]);
    ksum[t] = s;
  }
  __syncthreads();

  if (t < 196) {                               // z per row
    float s = 0.f;
    for (int d = 0; d < 64; ++d) s += b2f(qhd[t * 64 + d]) * ksum[d];
    zr[t] = 1.f / (s + EPS_);
  }

  // ---- kv GEMM: 4x4 tiles, wave handles tiles 2w, 2w+1 ----
  v8f zero = {};
  v8f ka[2]; ka[0] = zero; ka[1] = zero;
  const int tid0 = w * 2;
  for (int kk = 0; kk < 7; ++kk) {
    for (int ti = 0; ti < 2; ++ti) {
      int tile = tid0 + ti, mt = tile >> 2, nt = tile & 3;
      Frag a, b;
      int c = mt * 16 + ln;                    // A = k^T : A[c][j] = lk[j][c]
      int n = nt * 16 + ln;                    // B = v   : B[j][n]
      int kbs = half * 8, kb2 = half * 16;
      if (kk < 6) {                            // rows 0..191 always in range
        for (int e = 0; e < 16; ++e) {
          int j = kk * 32 + kb2 + e;
          a.s[e] = lk[j * 64 + c];
        }
        for (int e = 0; e < 16; ++e)
          b.s[e] = lv[(kk * 32 + kb2 + e) * 64 + n];
      } else {                                 // tail chunk: rows 192..223
        for (int e = 0; e < 16; ++e) {
          int j = 192 + kb2 + e;
          a.s[e] = (j < 196) ? lk[j * 64 + c] : (u16)0;
        }
        for (int e = 0; e < 16; ++e) {
          int j = 192 + kb2 + e;
          b.s[e] = (j < 196) ? lv[j * 64 + n] : (u16)0;
        }
      }
      ka[ti] = wmma_bf16(a, b, ka[ti]);
    }
  }
  for (int ti = 0; ti < 2; ++ti) {             // store kv transposed as bf16
    int tile = tid0 + ti, mt = tile >> 2, nt = tile & 3;
    for (int r = 0; r < 8; ++r) {
      int c = mt * 16 + r + half * 8;
      int d2 = nt * 16 + ln;
      kvT[d2 * 64 + c] = f2b(ka[ti][r]);
    }
  }
  __syncthreads();

  // ---- depthwise 5x5 conv on v -> ffm (overwrites lk; kv reads done) ----
  for (int e = t; e < 196 * 64; e += 256) {
    int ntok = e >> 6, d = e & 63;
    int y = ntok / 14, x0 = ntok % 14;
    float s = dwc_b[d];
    for (int ky = 0; ky < 5; ++ky) {
      int yy = y + ky - 2;
      if (yy < 0 || yy >= 14) continue;
      for (int kx = 0; kx < 5; ++kx) {
        int xx = x0 + kx - 2;
        if (xx < 0 || xx >= 14) continue;
        s += dwc_w[d * 25 + ky * 5 + kx] * b2f(lv[(yy * 14 + xx) * 64 + d]);
      }
    }
    ffm[e] = f2b(s);
  }
  __syncthreads();

  // ---- out GEMM: 13(M) x 4(N) tiles over 8 waves; A (q) from global ----
  for (int tile = w; tile < 52; tile += 8) {
    int mt = tile % 13, nt = tile / 13;
    v8f acc = zero;
    for (int kk = 0; kk < 2; ++kk) {
      Frag a, b;
      int m = mt * 16 + ln;                    // token row
      int kbs = half * 8, kb2 = half * 16;
      if (mt < 12) {                           // contiguous 8-elt runs: b128
        const uint4* qv4 = (const uint4*)&qhd[m * 64 + kk * 32 + kbs];
        uint4 g0 = qv4[0];                     // K = kk*32+kbs   .. +7
        uint4 g1 = qv4[2];                     // K = kk*32+16+kbs.. +7
        a.u[0] = g0.x; a.u[1] = g0.y; a.u[2] = g0.z; a.u[3] = g0.w;
        a.u[4] = g1.x; a.u[5] = g1.y; a.u[6] = g1.z; a.u[7] = g1.w;
      } else {
        for (int e = 0; e < 16; ++e) {
          int k = kk * 32 + ((e >> 3) << 4) + kbs + (e & 7);
          a.s[e] = (m < 196) ? qhd[m * 64 + k] : (u16)0;
        }
      }
      int n = nt * 16 + ln;
      const uint4* kv4 = (const uint4*)&kvT[n * 64 + kk * 32 + kbs];
      uint4 b0 = kv4[0], b1 = kv4[2];          // K = kbs..+7, 16+kbs..+7
      b.u[0] = b0.x; b.u[1] = b0.y; b.u[2] = b0.z; b.u[3] = b0.w;
      b.u[4] = b1.x; b.u[5] = b1.y; b.u[6] = b1.z; b.u[7] = b1.w;
      acc = wmma_bf16(a, b, acc);
    }
    for (int r = 0; r < 8; ++r) {
      int i = mt * 16 + r + half * 8;
      if (mt < 12 || i < 196) {
        int d = nt * 16 + ln;
        float val = acc[r] * zr[i] + b2f(ffm[i * 64 + d]);
        attn[(size_t)(bi * 196 + i) * 768 + h * 64 + d] = f2b(val);
      }
    }
  }
}

// ---------------------------------------------------------------------------
// Kernel D: out(50176x768) = attn @ Wp^T + bp  (fp32 output, double-buffered)
// ---------------------------------------------------------------------------
__global__ __launch_bounds__(256) void k_gemm_out(
    const u16* __restrict__ attn, const float* __restrict__ Wp,
    const float* __restrict__ bp, float* __restrict__ out)
{
  __shared__ __align__(16) u16 lA[2][128 * 32];
  __shared__ __align__(16) u16 lB[2][64 * 32];
  const int t = threadIdx.x;
  const int w = t >> 5, lane = t & 31, half = lane >> 4, ln = lane & 15;
  const int wm = w >> 1, wn = w & 1;
  const int mblk = blockIdx.x * 128;
  const int nblk = blockIdx.y * 64;

  auto load_tiles = [&](int kk, int buf) {
    const int k0 = kk * 32;
    for (int idx = t; idx < 512; idx += 256) {           // A: bf16 b128 copy
      int r = idx >> 2, c = (idx & 3) << 3;
      *(uint4*)&lA[buf][r * 32 + c] =
          *(const uint4*)&attn[(size_t)(mblk + r) * 768 + k0 + c];
    }
    for (int idx = t; idx < 512; idx += 256) {           // B: f32->bf16 pack
      int n = idx >> 3, c = (idx & 7) << 2;
      float4 f = *(const float4*)&Wp[(size_t)(nblk + n) * 768 + k0 + c];
      *(uint2*)&lB[buf][n * 32 + c] = make_uint2(pack2(f.x, f.y), pack2(f.z, f.w));
    }
  };

  v8f zero = {};
  v8f acc[2][2];
  for (int i = 0; i < 2; ++i) for (int j = 0; j < 2; ++j) acc[i][j] = zero;

  load_tiles(0, 0);
  for (int kk = 0; kk < 24; ++kk) {
    __syncthreads();
    if (kk + 1 < 24) load_tiles(kk + 1, (kk + 1) & 1);
    const u16* A_ = lA[kk & 1];
    const u16* B_ = lB[kk & 1];

    Frag a[2], b[2];
    for (int mt = 0; mt < 2; ++mt) {
      int m = wm * 32 + mt * 16 + ln;
      int kbs = half * 16;
      for (int i = 0; i < 8; ++i)
        a[mt].u[i] = *(const u32*)&A_[m * 32 + kbs + 2 * i];
    }
    for (int nt = 0; nt < 2; ++nt) {
      int n = wn * 32 + nt * 16 + ln;
      int kbs = half * 16;
      for (int i = 0; i < 8; ++i)
        b[nt].u[i] = *(const u32*)&B_[n * 32 + kbs + 2 * i];
    }
    for (int mt = 0; mt < 2; ++mt)
      for (int nt = 0; nt < 2; ++nt)
        acc[mt][nt] = wmma_bf16(a[mt], b[nt], acc[mt][nt]);
  }

  for (int mt = 0; mt < 2; ++mt)
    for (int nt = 0; nt < 2; ++nt)
      for (int r = 0; r < 8; ++r) {
        int m_g = mblk + wm * 32 + mt * 16 + r + half * 8;
        int n_g = nblk + wn * 32 + nt * 16 + ln;
        out[(size_t)m_g * 768 + n_g] = acc[mt][nt][r] + bp[n_g];
      }
}

// ---------------------------------------------------------------------------
extern "C" void kernel_launch(void* const* d_in, const int* in_sizes, int n_in,
                              void* d_out, int out_size, void* d_ws, size_t ws_size,
                              hipStream_t stream)
{
  const float* x       = (const float*)d_in[0];
  const float* Wq      = (const float*)d_in[1];
  const float* bq      = (const float*)d_in[2];
  const float* Wk      = (const float*)d_in[3];
  const float* bk      = (const float*)d_in[4];
  const float* Wv      = (const float*)d_in[5];
  const float* bv      = (const float*)d_in[6];
  const float* Wp      = (const float*)d_in[7];
  const float* bp      = (const float*)d_in[8];
  const float* scale_p = (const float*)d_in[9];
  const float* pos     = (const float*)d_in[10];
  const float* dwc_w   = (const float*)d_in[11];
  const float* dwc_b   = (const float*)d_in[12];

  const size_t SZ = (size_t)50176 * 768;      // elements per (M,C) buffer
  u16* qb = (u16*)d_ws;                       // bf16 q (row-major)
  u16* kb = qb + SZ;                          // bf16 k (row-major)
  u16* vh = kb + SZ;                          // bf16 v (head-split)
  u16* qh = vh + SZ;                          // bf16 focused q (head-split)
  u16* kh = qh + SZ;                          // bf16 focused k (head-split)
  u16* at = kh + SZ;                          // bf16 attn+conv (row-major)

  k_gemm_qkv<<<dim3(392, 36), 256, 0, stream>>>(x, Wq, bq, Wk, bk, Wv, bv,
                                                scale_p, pos, qb, kb, vh);
  k_focus<<<50176, 256, 0, stream>>>(qb, kb, qh, kh);
  k_attn_head<<<3072, 256, 0, stream>>>(qh, kh, vh, dwc_w, dwc_b, at);
  k_gemm_out<<<dim3(392, 12), 256, 0, stream>>>(at, Wp, bp, (float*)d_out);
}
